// NATLayer_21629455303333
// MI455X (gfx1250) — compile-verified, matrix operand order checked by reading, not run on hardware
//
#include <hip/hip_runtime.h>
#include <hip/hip_bf16.h>
#include <math.h>

typedef __attribute__((ext_vector_type(16))) _Float16 v16h;
typedef __attribute__((ext_vector_type(8)))  _Float16 v8h;
typedef __attribute__((ext_vector_type(8)))  float    v8f;

typedef unsigned int u32x4_t __attribute__((ext_vector_type(4)));
typedef int          i32x8_t __attribute__((ext_vector_type(8)));
typedef int          i32x4_t __attribute__((ext_vector_type(4)));

#define B_    16
#define C_    392
#define CP_   416      // C padded to multiple of 32 (K padding for WMMA)
#define HW_   4096
#define C3_   1176     // 3*C
#define HD_   1568     // hidden = 4*C, already multiple of 32
#define NH_   8
#define HDIM_ 49

union Frag { v16h v; v8h h[2]; };

__device__ __forceinline__ v16h make_frag(const _Float16* p) {
  // 16-bit fragment layout (ISA 7.12.2): per lane, VGPRs 0-3 hold K = hi*8..hi*8+7,
  // VGPRs 4-7 hold K = 16+hi*8..16+hi*8+7  ->  two contiguous 16B LDS loads.
  Frag f;
  f.h[0] = *(const v8h*)(p);
  f.h[1] = *(const v8h*)(p + 16);
  return f.v;
}

// Issue a TDM 2D tile load: [rows=128][cols=32 halves] from global (row stride
// KP halves) into LDS at lds_addr, with LDS padding 16B after every 64B so the
// landed layout has an 80B (40-half) row stride = the bank-conflict-free layout.
__device__ __forceinline__ void tdm_load_a_tile(const _Float16* gptr,
                                                unsigned lds_addr, int KP) {
  unsigned long long ga = (unsigned long long)(const void*)gptr;
  u32x4_t g0;
  g0.x = 1u;                                   // count=1, is_restore=0, gather=0
  g0.y = lds_addr;                             // LDS byte address
  g0.z = (unsigned)ga;                         // global_addr[31:0]
  g0.w = (unsigned)((ga >> 32) & 0x1FFFFFFu)   // global_addr[56:32]
         | (2u << 30);                         // type=2 ("image")
  i32x8_t g1;
  // dword0: wg_mask=0 | data_size(2B)=1<<16 | pad_enable=1<<20
  //         | pad_interval(16 DW)=3<<22 | pad_amount(4 DW)=3<<25
  g1.s0 = (int)((1u << 16) | (1u << 20) | (3u << 22) | (3u << 25));
  g1.s1 = (int)(32u << 16);                    // tensor_dim0[15:0] (bits 63:48)
  g1.s2 = (int)(128u << 16);                   // tensor_dim1[15:0] (bits 95:80)
  g1.s3 = (int)(32u << 16);                    // tile_dim0 (bits 127:112)
  g1.s4 = 128;                                 // tile_dim1 (bits 143:128)
  g1.s5 = KP;                                  // tensor_dim0_stride[31:0]
  g1.s6 = 0;                                   // stride hi / dim1_stride lo
  g1.s7 = 0;
  i32x4_t gz = {0, 0, 0, 0};
#if defined(__clang_major__) && (__clang_major__ >= 23)
  i32x8_t gz8 = {0, 0, 0, 0, 0, 0, 0, 0};
  __builtin_amdgcn_tensor_load_to_lds(g0, g1, gz, gz, gz8, 0);
#else
  __builtin_amdgcn_tensor_load_to_lds(g0, g1, gz, gz, 0);
#endif
}

// ---------------------------------------------------------------------------
// Weight convert + pad: f32 [M][K] -> f16 [MP][KP], zero-padded
// ---------------------------------------------------------------------------
__global__ __launch_bounds__(256)
void cvt_pad_kernel(const float* __restrict__ src, _Float16* __restrict__ dst,
                    int M, int K, int KP, int total) {
  int t = blockIdx.x * 256 + threadIdx.x;
  if (t >= total) return;
  int m = t / KP, k = t - m * KP;
  dst[t] = (m < M && k < K) ? (_Float16)src[(size_t)m * K + k] : (_Float16)0.f;
}

// ---------------------------------------------------------------------------
// Channel LayerNorm (per pixel over C), f32 in -> f16 out [B][CP][HW], K rows
// C..CP-1 zeroed so GEMM needs no K guards.
// ---------------------------------------------------------------------------
__global__ __launch_bounds__(256)
void ln_kernel(const float* __restrict__ x, const float* __restrict__ g,
               const float* __restrict__ bta, _Float16* __restrict__ out) {
  int p   = blockIdx.x * 256 + threadIdx.x;   // 0 .. B*HW-1
  int b   = p >> 12;
  int pix = p & (HW_ - 1);
  size_t base = (size_t)b * C_ * HW_ + pix;
  float sum = 0.f, sq = 0.f;
  for (int c = 0; c < C_; ++c) {
    float v = x[base + (size_t)c * HW_];
    sum += v; sq += v * v;
  }
  float mu  = sum * (1.0f / C_);
  float var = sq * (1.0f / C_) - mu * mu;
  float rs  = rsqrtf(var + 1e-5f);
  size_t ob = (size_t)b * CP_ * HW_ + pix;
  for (int c = 0; c < C_; ++c) {
    float v = x[base + (size_t)c * HW_];
    out[ob + (size_t)c * HW_] = (_Float16)(((v - mu) * rs) * g[c] + bta[c]);
  }
  for (int c = C_; c < CP_; ++c) out[ob + (size_t)c * HW_] = (_Float16)0.f;
}

// ---------------------------------------------------------------------------
// Zero the K-pad rows of a [B][CP][HW] f16 buffer (rows C..CP-1)
// ---------------------------------------------------------------------------
__global__ __launch_bounds__(256)
void zero_pad_kernel(_Float16* __restrict__ buf) {
  int t = blockIdx.x * 256 + threadIdx.x;     // B * (CP-C) * HW
  int p = t & (HW_ - 1);
  int r = (t >> 12) % (CP_ - C_);
  int b = t / ((CP_ - C_) * HW_);
  buf[(size_t)b * CP_ * HW_ + (size_t)(C_ + r) * HW_ + p] = (_Float16)0.f;
}

// ---------------------------------------------------------------------------
// Per-pixel "attention": softmax over 49 head-channels of q*scale*k + bias,
// times v, then the reference's row-major reshape scatter into [B][CP][HW].
// ---------------------------------------------------------------------------
__global__ __launch_bounds__(256)
void attn_kernel(const _Float16* __restrict__ qkv, const float* __restrict__ rbias,
                 _Float16* __restrict__ outp) {
  int t    = blockIdx.x * 256 + threadIdx.x;  // B*NH*HW threads
  int pix  = t & (HW_ - 1);
  int head = (t >> 12) & (NH_ - 1);
  int b    = t >> 15;
  size_t qb = ((size_t)b * C3_ + (size_t)head * HDIM_) * HW_ + pix;
  size_t kb = qb + (size_t)C_ * HW_;
  size_t vb = kb + (size_t)C_ * HW_;
  const float scale = 1.0f / 7.0f;            // 49^-0.5
  float s[HDIM_];
  float mx = -3.0e38f;
#pragma unroll
  for (int d = 0; d < HDIM_; ++d) {
    float q  = (float)qkv[qb + (size_t)d * HW_];
    float k  = (float)qkv[kb + (size_t)d * HW_];
    float sv = q * scale * k + rbias[head * HDIM_ + d];
    s[d] = sv;
    mx = fmaxf(mx, sv);
  }
  float sum = 0.f;
#pragma unroll
  for (int d = 0; d < HDIM_; ++d) { s[d] = expf(s[d] - mx); sum += s[d]; }
  float inv = 1.0f / sum;
  size_t obase = (size_t)b * CP_ * HW_;
#pragma unroll
  for (int d = 0; d < HDIM_; ++d) {
    float v = (float)qkv[vb + (size_t)d * HW_];
    float o = s[d] * inv * v;
    size_t flat = ((size_t)head * HW_ + pix) * HDIM_ + d;   // [n,HW,hd] flatten
    size_t c  = flat >> 12;                                 // / HW
    size_t p2 = flat & (HW_ - 1);
    outp[obase + c * HW_ + p2] = (_Float16)o;
  }
}

// ---------------------------------------------------------------------------
// WMMA GEMM: D[M,N=HW] = A[M,K] * Bsrc[K,HW]  (per batch via blockIdx.z)
// Block tile 128x64, 8 waves, each wave 2x2 v_wmma_f32_16x16x32_f16 tiles.
// Double-buffered software pipeline: TDM stages A tile k+1 (hardware LDS
// padding) and all waves stage B tile k+1 (transposed) while tile k is being
// multiplied; TENSORcnt is drained only at the bottom barrier.
// EPI 0: f16 out = acc + bias
// EPI 1: f32 out = acc + bias + residual
// EPI 2: f16 out = gelu_erf(acc + bias)
// ---------------------------------------------------------------------------
template<int EPI>
__global__ __launch_bounds__(256)
void gemm_wmma_kernel(const _Float16* __restrict__ A,     // [MP][KP], zero-padded
                      const _Float16* __restrict__ Bsrc,  // [B][>=K][HW] f16
                      const float* __restrict__ bias,     // [M]
                      const float* __restrict__ res,      // [B][M][HW] (EPI==1)
                      _Float16* __restrict__ out16,
                      float* __restrict__ out32,
                      int M, int KP,
                      size_t b_bs, size_t out_bs, size_t res_bs) {
  __shared__ _Float16 As[2][128 * 40];  // [m][k], stride 40 halves (80B rows)
  __shared__ _Float16 Bs[2][64 * 40];   // transposed [n][k], stride 40 halves

  const int t    = threadIdx.x;
  const int lane = t & 31;
  const int hi   = lane >> 4;
  const int lr   = lane & 15;
  const int w    = t >> 5;
  const int wm   = (w & 3) * 32;      // 4 waves along M
  const int wn   = (w >> 2) * 32;     // 2 waves along N

  const int n0 = blockIdx.x * 64;
  const int m0 = blockIdx.y * 128;
  const int bb = blockIdx.z;

  const _Float16* Bp = Bsrc + (size_t)bb * b_bs;

  v8f acc[2][2] = {};

  const int bk = t >> 3;              // 0..31 : k within tile
  const int bn = (t & 7) << 3;        // 0..56 : n group of 8

  // ---- prologue: bring tile 0 into buffer 0 ----
  if (t < 32) {
    tdm_load_a_tile(A + (size_t)m0 * KP,
                    (unsigned)(uintptr_t)(void*)&As[0][0], KP);
  }
  {
    v8h bv = *(const v8h*)(Bp + (size_t)bk * HW_ + (n0 + bn));
#pragma unroll
    for (int i = 0; i < 8; ++i) Bs[0][(bn + i) * 40 + bk] = bv[i];
  }
  if (t < 32) __builtin_amdgcn_s_wait_tensorcnt((short)0);
  __syncthreads();

  // ---- steady state: stage tile k+1 while computing tile k ----
  int buf = 0;
  for (int k0 = 0; k0 < KP; k0 += 32) {
    const int nxt = buf ^ 1;
    if (k0 + 32 < KP) {
      if (t < 32) {
        tdm_load_a_tile(A + (size_t)m0 * KP + (k0 + 32),
                        (unsigned)(uintptr_t)(void*)&As[nxt][0], KP);
      }
      v8h bv = *(const v8h*)(Bp + (size_t)(k0 + 32 + bk) * HW_ + (n0 + bn));
#pragma unroll
      for (int i = 0; i < 8; ++i) Bs[nxt][(bn + i) * 40 + bk] = bv[i];
      if (k0 + 64 < KP) {             // hint tile k+2 -> global_prefetch_b8
        __builtin_prefetch(Bp + (size_t)(k0 + 64 + bk) * HW_ + (n0 + bn), 0, 1);
      }
    }

    v16h af[2], bf[2];
#pragma unroll
    for (int i = 0; i < 2; ++i) {
      af[i] = make_frag(&As[buf][(wm + i * 16 + lr) * 40 + hi * 8]);
      bf[i] = make_frag(&Bs[buf][(wn + i * 16 + lr) * 40 + hi * 8]);
    }
#pragma unroll
    for (int i = 0; i < 2; ++i)
#pragma unroll
      for (int j = 0; j < 2; ++j)
        acc[i][j] = __builtin_amdgcn_wmma_f32_16x16x32_f16(
            false, af[i], false, bf[j], (short)0, acc[i][j], false, false);

    // tile k+1's DMA has had the whole compute phase to land; drain and swap
    if (t < 32) __builtin_amdgcn_s_wait_tensorcnt((short)0);
    __syncthreads();
    buf = nxt;
  }

  // epilogue: D layout (ISA 7.12.2): element r -> row = r + 8*hi, col = lane&15
#pragma unroll
  for (int i = 0; i < 2; ++i) {
#pragma unroll
    for (int j = 0; j < 2; ++j) {
      int n = n0 + wn + j * 16 + lr;
#pragma unroll
      for (int r = 0; r < 8; ++r) {
        int m = m0 + wm + i * 16 + r + hi * 8;
        if (m < M) {
          float v = acc[i][j][r] + bias[m];
          size_t oidx = (size_t)bb * out_bs + (size_t)m * HW_ + n;
          if (EPI == 0) {
            out16[oidx] = (_Float16)v;
          } else if (EPI == 1) {
            out32[oidx] = v + res[(size_t)bb * res_bs + (size_t)m * HW_ + n];
          } else {
            float g = 0.5f * v * (1.0f + erff(v * 0.7071067811865475f));
            out16[oidx] = (_Float16)g;
          }
        }
      }
    }
  }
}

// ---------------------------------------------------------------------------
extern "C" void kernel_launch(void* const* d_in, const int* in_sizes, int n_in,
                              void* d_out, int out_size, void* d_ws, size_t ws_size,
                              hipStream_t stream) {
  (void)in_sizes; (void)n_in; (void)out_size; (void)ws_size;

  const float* x      = (const float*)d_in[0];
  const float* ln1_g  = (const float*)d_in[1];
  const float* ln1_b  = (const float*)d_in[2];
  const float* qkv_w  = (const float*)d_in[3];
  const float* qkv_b  = (const float*)d_in[4];
  const float* rbias  = (const float*)d_in[5];
  const float* proj_w = (const float*)d_in[6];
  const float* proj_b = (const float*)d_in[7];
  const float* ln2_g  = (const float*)d_in[8];
  const float* ln2_b  = (const float*)d_in[9];
  const float* fc1_w  = (const float*)d_in[10];
  const float* fc1_b  = (const float*)d_in[11];
  const float* fc2_w  = (const float*)d_in[12];
  const float* fc2_b  = (const float*)d_in[13];
  float* out = (float*)d_out;

  // workspace carve-up
  char* ws = (char*)d_ws;
  size_t off = 0;
  auto al = [&](size_t bytes) { size_t o = off; off += (bytes + 255) & ~(size_t)255; return o; };
  _Float16* qkvw16  = (_Float16*)(ws + al((size_t)1280 * 416 * 2));
  _Float16* projw16 = (_Float16*)(ws + al((size_t)512  * 416 * 2));
  _Float16* fc1w16  = (_Float16*)(ws + al((size_t)1664 * 416 * 2));
  _Float16* fc2w16  = (_Float16*)(ws + al((size_t)512  * 1568 * 2));
  _Float16* h16     = (_Float16*)(ws + al((size_t)B_ * CP_ * HW_ * 2));
  _Float16* qkv16   = (_Float16*)(ws + al((size_t)B_ * C3_ * HW_ * 2));
  _Float16* ao16    = (_Float16*)(ws + al((size_t)B_ * CP_ * HW_ * 2));
  float*    yres    = (float*)   (ws + al((size_t)B_ * C_  * HW_ * 4));
  _Float16* h2_16   = (_Float16*)(ws + al((size_t)B_ * CP_ * HW_ * 2));
  _Float16* act16   = (_Float16*)(ws + al((size_t)B_ * HD_ * HW_ * 2));

  // 1) weights -> padded f16
  cvt_pad_kernel<<<(1280 * 416) / 256, 256, 0, stream>>>(qkv_w,  qkvw16,  C3_, C_,  416,  1280 * 416);
  cvt_pad_kernel<<<(512  * 416) / 256, 256, 0, stream>>>(proj_w, projw16, C_,  C_,  416,  512  * 416);
  cvt_pad_kernel<<<(1664 * 416) / 256, 256, 0, stream>>>(fc1_w,  fc1w16,  HD_, C_,  416,  1664 * 416);
  cvt_pad_kernel<<<(512 * 1568) / 256, 256, 0, stream>>>(fc2_w,  fc2w16,  C_,  HD_, 1568, 512 * 1568);

  // 2) LN1 -> h16
  ln_kernel<<<(B_ * HW_) / 256, 256, 0, stream>>>(x, ln1_g, ln1_b, h16);

  // 3) qkv GEMM: [1176 x 392] * [392 x 4096] per batch -> qkv16 (f16, +bias)
  gemm_wmma_kernel<0><<<dim3(HW_ / 64, 1280 / 128, B_), 256, 0, stream>>>(
      qkvw16, h16, qkv_b, nullptr, qkv16, nullptr,
      C3_, CP_, (size_t)CP_ * HW_, (size_t)C3_ * HW_, 0);

  // 4) attention (elementwise softmax over head channels) -> ao16 (scattered)
  zero_pad_kernel<<<(B_ * (CP_ - C_) * HW_) / 256, 256, 0, stream>>>(ao16);
  attn_kernel<<<(B_ * NH_ * HW_) / 256, 256, 0, stream>>>(qkv16, rbias, ao16);

  // 5) proj GEMM + bias + residual(x) -> yres (f32)
  gemm_wmma_kernel<1><<<dim3(HW_ / 64, 512 / 128, B_), 256, 0, stream>>>(
      projw16, ao16, proj_b, x, nullptr, yres,
      C_, CP_, (size_t)CP_ * HW_, (size_t)C_ * HW_, (size_t)C_ * HW_);

  // 6) LN2 -> h2_16
  ln_kernel<<<(B_ * HW_) / 256, 256, 0, stream>>>(yres, ln2_g, ln2_b, h2_16);

  // 7) fc1 GEMM + bias + exact GELU -> act16 (f16)
  gemm_wmma_kernel<2><<<dim3(HW_ / 64, 1664 / 128, B_), 256, 0, stream>>>(
      fc1w16, h2_16, fc1_b, nullptr, act16, nullptr,
      HD_, CP_, (size_t)CP_ * HW_, (size_t)HD_ * HW_, 0);

  // 8) fc2 GEMM + bias + residual(yres) -> d_out (f32)
  gemm_wmma_kernel<1><<<dim3(HW_ / 64, 512 / 128, B_), 256, 0, stream>>>(
      fc2w16, act16, fc2_b, yres, nullptr, out,
      C_, HD_, (size_t)HD_ * HW_, (size_t)C_ * HW_, (size_t)C_ * HW_);
}
